// BottleneckTransformer_37254546325612
// MI455X (gfx1250) — matrix operand order repaired and checked by reading, not compiled, not run on hardware
//
#include <hip/hip_runtime.h>
#include <hip/hip_bf16.h>

typedef __attribute__((ext_vector_type(16))) _Float16 v16h;
typedef __attribute__((ext_vector_type(8)))  float    v8f;

#define DIMC 256
#define SEQ  1024
#define BATCH 16
#define NHEAD 8
#define HDIM 32
#define NTOK (BATCH * SEQ)   /* 16384 */
#define FFND 1024

// ---------------------------------------------------------------------------
// WMMA helpers (CDNA5 wave32, 16x16x32 f16 -> f32)
// ---------------------------------------------------------------------------
static __device__ __forceinline__ v8f wmma16(v16h a, v16h b, v8f c) {
  return __builtin_amdgcn_wmma_f32_16x16x32_f16(false, a, false, b, (short)0, c,
                                                false, false);
}

// A fragment: 16x32 f16, row-major source with leading dim lda (halves).
// Lanes 0-15: row M=lane, K in {0..7,16..23}; lanes 16-31: row M=lane-16,
// K in {8..15,24..31}.
static __device__ __forceinline__ v16h load_A(const _Float16* p, int lda) {
  const int lane = threadIdx.x & 31;
  const _Float16* r = p + (lane & 15) * lda + ((lane >> 4) << 3);
  v16h a;
#pragma unroll
  for (int v = 0; v < 4; ++v) {
    a[2 * v]     = r[2 * v];
    a[2 * v + 1] = r[2 * v + 1];
    a[8 + 2 * v] = r[16 + 2 * v];
    a[9 + 2 * v] = r[17 + 2 * v];
  }
  return a;
}

// B fragment: 32x16 f16 from an N-major source (element [K][N] at p[N*ldb+K]).
// Lane = N (mod 16); lanes 0-15 hold K in {0..7,16..23}, lanes 16-31 K in {8..15,24..31}.
static __device__ __forceinline__ v16h load_B(const _Float16* p, int ldb) {
  const int lane = threadIdx.x & 31;
  const _Float16* r = p + (lane & 15) * ldb + ((lane >> 4) << 3);
  v16h b;
#pragma unroll
  for (int i = 0; i < 16; ++i) b[i] = r[i < 8 ? i : i + 8];
  return b;
}

// Software-pipelined wave tile 16(M) x 64(N): 1 A-frag x 4 B-frags -> 4 WMMAs
// per k-step; the next k-step's 5 fragments are prefetched into fresh
// registers before the current WMMAs, so waits don't drain loadcnt to zero.
// Register footprint (~112 VGPRs of fragments+accs) stays spill-free.
static __device__ __forceinline__ void gemm16x64(const _Float16* __restrict__ A,
                                                 int lda,
                                                 const _Float16* __restrict__ W,
                                                 int ldw, int K, v8f acc[4]) {
  v16h a  = load_A(A, lda);
  v16h b0 = load_B(W, ldw);
  v16h b1 = load_B(W + 16 * ldw, ldw);
  v16h b2 = load_B(W + 32 * ldw, ldw);
  v16h b3 = load_B(W + 48 * ldw, ldw);
  for (int k0 = 32; k0 < K; k0 += 32) {
    v16h na  = load_A(A + k0, lda);
    v16h nb0 = load_B(W + k0, ldw);
    v16h nb1 = load_B(W + 16 * ldw + k0, ldw);
    v16h nb2 = load_B(W + 32 * ldw + k0, ldw);
    v16h nb3 = load_B(W + 48 * ldw + k0, ldw);
    acc[0] = wmma16(a, b0, acc[0]);
    acc[1] = wmma16(a, b1, acc[1]);
    acc[2] = wmma16(a, b2, acc[2]);
    acc[3] = wmma16(a, b3, acc[3]);
    a = na; b0 = nb0; b1 = nb1; b2 = nb2; b3 = nb3;
  }
  acc[0] = wmma16(a, b0, acc[0]);
  acc[1] = wmma16(a, b1, acc[1]);
  acc[2] = wmma16(a, b2, acc[2]);
  acc[3] = wmma16(a, b3, acc[3]);
}

// ---------------------------------------------------------------------------
// Elementwise / LN kernels
// ---------------------------------------------------------------------------
__global__ void k_f32_to_f16(const float* __restrict__ in,
                             _Float16* __restrict__ out, int n) {
  int i = blockIdx.x * 256 + threadIdx.x;
  if (i < n) out[i] = (_Float16)in[i];
}

// LN over channels for x stored [b][c][s]; write x_ln f16 row-major [tok][c].
__global__ void k_ln1(const float* __restrict__ x, const float* __restrict__ g,
                      const float* __restrict__ be, _Float16* __restrict__ xln) {
  int tok = blockIdx.x * 256 + threadIdx.x;      // coalesced across s
  int b = tok >> 10, s = tok & 1023;
  const float* xb = x + ((size_t)b * DIMC) * SEQ + s;
  float sum = 0.f, sq = 0.f;
  for (int c = 0; c < DIMC; ++c) {
    float v = xb[(size_t)c * SEQ];
    sum += v; sq += v * v;
  }
  float mu = sum * (1.f / DIMC);
  float var = sq * (1.f / DIMC) - mu * mu;
  float rs = rsqrtf(var + 1e-5f);
  _Float16* o = xln + (size_t)tok * DIMC;
  for (int c = 0; c < DIMC; ++c) {
    float v = xb[(size_t)c * SEQ];
    o[c] = (_Float16)(((v - mu) * rs) * g[c] + be[c]);
  }
}

// LN over row-major y [tok][256]: one wave per token.
__global__ void k_ln2(const float* __restrict__ y, const float* __restrict__ g,
                      const float* __restrict__ be, _Float16* __restrict__ out) {
  int wv = threadIdx.x >> 5, lane = threadIdx.x & 31;
  int tok = blockIdx.x * 8 + wv;
  const float* r = y + (size_t)tok * DIMC;
  float vals[8], sum = 0.f, sq = 0.f;
#pragma unroll
  for (int i = 0; i < 8; ++i) {
    float v = r[lane + i * 32];
    vals[i] = v; sum += v; sq += v * v;
  }
#pragma unroll
  for (int m = 1; m < 32; m <<= 1) {
    sum += __shfl_xor(sum, m, 32);
    sq  += __shfl_xor(sq,  m, 32);
  }
  float mu = sum * (1.f / DIMC);
  float rs = rsqrtf(sq * (1.f / DIMC) - mu * mu + 1e-5f);
  _Float16* o = out + (size_t)tok * DIMC;
#pragma unroll
  for (int i = 0; i < 8; ++i) {
    int c = lane + i * 32;
    o[c] = (_Float16)(((vals[i] - mu) * rs) * g[c] + be[c]);
  }
}

// ---------------------------------------------------------------------------
// GEMM kernels (256 threads = 8 waves; block tile 32(M) x 256(N))
// wave w: rows = blockIdx.y*32 + (w>>2)*16, cols = blockIdx.x*256 + (w&3)*64
// ---------------------------------------------------------------------------
__global__ void k_qkv(const _Float16* __restrict__ xln,
                      const _Float16* __restrict__ wqkv,
                      _Float16* __restrict__ qb, _Float16* __restrict__ kb,
                      _Float16* __restrict__ vT) {
  int wv = threadIdx.x >> 5, lane = threadIdx.x & 31;
  int row0 = blockIdx.y * 32 + (wv >> 2) * 16;
  int col0 = blockIdx.x * 256 + (wv & 3) * 64;
  v8f acc[4] = {};
  gemm16x64(xln + (size_t)row0 * DIMC, DIMC, wqkv + (size_t)col0 * DIMC, DIMC,
            DIMC, acc);
  const float qscale = 0.17677669529663687f;  // 1/sqrt(32)
#pragma unroll
  for (int n = 0; n < 4; ++n) {
    int col = col0 + n * 16 + (lane & 15);
#pragma unroll
    for (int r = 0; r < 8; ++r) {
      int row = row0 + r + ((lane >> 4) << 3);
      int b = row >> 10, sidx = row & 1023;
      float v = acc[n][r];
      if (col < 256) {               // Q, pre-scaled
        int hd = col >> 5, d = col & 31;
        qb[(((size_t)(b * NHEAD + hd)) * SEQ + sidx) * HDIM + d] =
            (_Float16)(v * qscale);
      } else if (col < 512) {        // K
        int j = col - 256, hd = j >> 5, d = j & 31;
        kb[(((size_t)(b * NHEAD + hd)) * SEQ + sidx) * HDIM + d] = (_Float16)v;
      } else {                       // V transposed: [bh][d][s]
        int j = col - 512, hd = j >> 5, d = j & 31;
        vT[(((size_t)(b * NHEAD + hd)) * HDIM + d) * SEQ + sidx] = (_Float16)v;
      }
    }
  }
}

__global__ void k_proj(const _Float16* __restrict__ attn,
                       const _Float16* __restrict__ wp,
                       const float* __restrict__ bp,
                       const float* __restrict__ x, float* __restrict__ y) {
  int wv = threadIdx.x >> 5, lane = threadIdx.x & 31;
  int row0 = blockIdx.y * 32 + (wv >> 2) * 16;
  int col0 = blockIdx.x * 256 + (wv & 3) * 64;
  v8f acc[4] = {};
  gemm16x64(attn + (size_t)row0 * DIMC, DIMC, wp + (size_t)col0 * DIMC, DIMC,
            DIMC, acc);
#pragma unroll
  for (int n = 0; n < 4; ++n) {
    int col = col0 + n * 16 + (lane & 15);
#pragma unroll
    for (int r = 0; r < 8; ++r) {
      int row = row0 + r + ((lane >> 4) << 3);
      int b = row >> 10, sidx = row & 1023;
      float v = acc[n][r] + bp[col];
      y[(size_t)row * DIMC + col] =
          x[((size_t)b * DIMC + col) * SEQ + sidx] + v;  // residual
    }
  }
}

__global__ void k_ffn1(const _Float16* __restrict__ ln2,
                       const _Float16* __restrict__ w1,
                       const float* __restrict__ b1,
                       _Float16* __restrict__ h1) {
  int wv = threadIdx.x >> 5, lane = threadIdx.x & 31;
  int row0 = blockIdx.y * 32 + (wv >> 2) * 16;
  int col0 = blockIdx.x * 256 + (wv & 3) * 64;
  v8f acc[4] = {};
  gemm16x64(ln2 + (size_t)row0 * DIMC, DIMC, w1 + (size_t)col0 * DIMC, DIMC,
            DIMC, acc);
#pragma unroll
  for (int n = 0; n < 4; ++n) {
    int col = col0 + n * 16 + (lane & 15);
#pragma unroll
    for (int r = 0; r < 8; ++r) {
      int row = row0 + r + ((lane >> 4) << 3);
      float v = fmaxf(acc[n][r] + b1[col], 0.f);  // ReLU
      h1[(size_t)row * FFND + col] = (_Float16)v;
    }
  }
}

__global__ void k_ffn2(const _Float16* __restrict__ h1,
                       const _Float16* __restrict__ w2,
                       const float* __restrict__ b2,
                       const float* __restrict__ y, float* __restrict__ out) {
  int wv = threadIdx.x >> 5, lane = threadIdx.x & 31;
  int row0 = blockIdx.y * 32 + (wv >> 2) * 16;
  int col0 = blockIdx.x * 256 + (wv & 3) * 64;
  v8f acc[4] = {};
  gemm16x64(h1 + (size_t)row0 * FFND, FFND, w2 + (size_t)col0 * FFND, FFND,
            FFND, acc);
#pragma unroll
  for (int n = 0; n < 4; ++n) {
    int col = col0 + n * 16 + (lane & 15);
#pragma unroll
    for (int r = 0; r < 8; ++r) {
      int row = row0 + r + ((lane >> 4) << 3);
      int b = row >> 10, sidx = row & 1023;
      float v = acc[n][r] + b2[col];
      // final residual + transpose back to [b][c][s]
      out[((size_t)b * DIMC + col) * SEQ + sidx] =
          y[(size_t)row * DIMC + col] + v;
    }
  }
}

// ---------------------------------------------------------------------------
// Flash attention: one block per (b,h, 128-row q-tile); each wave owns 16 q
// rows. head_dim=32 == K of one 16x16x32 WMMA. k-tile = 64: 4 score WMMAs +
// 4 P*V WMMAs per iteration; softmax/LDS restripe amortized over 8 WMMAs.
// ---------------------------------------------------------------------------
__global__ void k_attn(const _Float16* __restrict__ qb,
                       const _Float16* __restrict__ kb,
                       const _Float16* __restrict__ vT,
                       _Float16* __restrict__ attn_out) {
  __shared__ _Float16 P[8][16][64];  // per-wave P tile (C-layout -> A-layout)
  int wv = threadIdx.x >> 5, lane = threadIdx.x & 31;
  int bh = blockIdx.x >> 3;          // 0..127 = b*8+h
  int qt = blockIdx.x & 7;
  int b = bh >> 3, h = bh & 7;
  int qrow = qt * 128 + wv * 16;     // row offset within sequence

  const _Float16* Q  = qb + ((size_t)bh * SEQ + qrow) * HDIM;
  const _Float16* Kp = kb + (size_t)bh * SEQ * HDIM;
  const _Float16* Vt = vT + (size_t)bh * HDIM * SEQ;

  v16h aq = load_A(Q, HDIM);  // whole head_dim fits one A fragment
  v8f o0 = {}, o1 = {};
  float mstat[8], lstat[8];
#pragma unroll
  for (int r = 0; r < 8; ++r) { mstat[r] = -3.0e38f; lstat[r] = 0.f; }

  for (int kt = 0; kt < SEQ; kt += 64) {
    v16h bk0 = load_B(Kp + (size_t)kt * HDIM, HDIM);
    v16h bk1 = load_B(Kp + (size_t)(kt + 16) * HDIM, HDIM);
    v16h bk2 = load_B(Kp + (size_t)(kt + 32) * HDIM, HDIM);
    v16h bk3 = load_B(Kp + (size_t)(kt + 48) * HDIM, HDIM);
    v8f z = {};
    v8f s0 = wmma16(aq, bk0, z);
    v8f s1 = wmma16(aq, bk1, z);
    v8f s2 = wmma16(aq, bk2, z);
    v8f s3 = wmma16(aq, bk3, z);
#pragma unroll
    for (int r = 0; r < 8; ++r) {
      float mx = fmaxf(fmaxf(s0[r], s1[r]), fmaxf(s2[r], s3[r]));
#pragma unroll
      for (int msk = 1; msk < 16; msk <<= 1)
        mx = fmaxf(mx, __shfl_xor(mx, msk, 32));   // row-max (16-lane half)
      float mnew = fmaxf(mstat[r], mx);
      float p0 = __expf(s0[r] - mnew);
      float p1 = __expf(s1[r] - mnew);
      float p2 = __expf(s2[r] - mnew);
      float p3 = __expf(s3[r] - mnew);
      float rs = (p0 + p1) + (p2 + p3);
#pragma unroll
      for (int msk = 1; msk < 16; msk <<= 1)
        rs += __shfl_xor(rs, msk, 32);             // row-sum
      float alpha = __expf(mstat[r] - mnew);
      mstat[r] = mnew;
      lstat[r] = lstat[r] * alpha + rs;
      o0[r] = o0[r] * alpha;
      o1[r] = o1[r] * alpha;
      int mr = r + ((lane >> 4) << 3);
      int c = lane & 15;
      P[wv][mr][c]      = (_Float16)p0;
      P[wv][mr][c + 16] = (_Float16)p1;
      P[wv][mr][c + 32] = (_Float16)p2;
      P[wv][mr][c + 48] = (_Float16)p3;
    }
    __syncthreads();
    v16h ap0 = load_A(&P[wv][0][0], 64);   // P cols 0..31
    v16h ap1 = load_A(&P[wv][0][32], 64);  // P cols 32..63
    __syncthreads();
    o0 = wmma16(ap0, load_B(Vt + 0 * SEQ + kt, SEQ), o0);        // d 0..15
    o0 = wmma16(ap1, load_B(Vt + 0 * SEQ + kt + 32, SEQ), o0);
    o1 = wmma16(ap0, load_B(Vt + 16 * SEQ + kt, SEQ), o1);       // d 16..31
    o1 = wmma16(ap1, load_B(Vt + 16 * SEQ + kt + 32, SEQ), o1);
  }

#pragma unroll
  for (int r = 0; r < 8; ++r) {
    int mr = r + ((lane >> 4) << 3);
    int row = b * SEQ + qrow + mr;                 // global token
    float inv = 1.f / lstat[r];
    attn_out[(size_t)row * DIMC + h * HDIM + (lane & 15)] =
        (_Float16)(o0[r] * inv);
    attn_out[(size_t)row * DIMC + h * HDIM + 16 + (lane & 15)] =
        (_Float16)(o1[r] * inv);
  }
}

// ---------------------------------------------------------------------------
// Host launcher
// ---------------------------------------------------------------------------
extern "C" void kernel_launch(void* const* d_in, const int* in_sizes, int n_in,
                              void* d_out, int out_size, void* d_ws,
                              size_t ws_size, hipStream_t stream) {
  (void)in_sizes; (void)n_in; (void)out_size; (void)ws_size;
  const float* x     = (const float*)d_in[0];
  const float* wqkv  = (const float*)d_in[1];
  const float* wproj = (const float*)d_in[2];
  const float* bproj = (const float*)d_in[3];
  const float* g1    = (const float*)d_in[4];
  const float* be1   = (const float*)d_in[5];
  const float* g2    = (const float*)d_in[6];
  const float* be2   = (const float*)d_in[7];
  const float* w1    = (const float*)d_in[8];
  const float* b1    = (const float*)d_in[9];
  const float* w2    = (const float*)d_in[10];
  const float* b2    = (const float*)d_in[11];
  float* out = (float*)d_out;

  char* ws = (char*)d_ws;
  auto carve = [&](size_t bytes) -> char* {
    char* p = ws;
    ws += (bytes + 255) & ~(size_t)255;
    return p;
  };
  _Float16* wqkv_h = (_Float16*)carve((size_t)768 * 256 * 2);
  _Float16* wprj_h = (_Float16*)carve((size_t)256 * 256 * 2);
  _Float16* w1_h   = (_Float16*)carve((size_t)1024 * 256 * 2);
  _Float16* w2_h   = (_Float16*)carve((size_t)256 * 1024 * 2);
  _Float16* xln    = (_Float16*)carve((size_t)NTOK * DIMC * 2);
  _Float16* qb     = (_Float16*)carve((size_t)BATCH * NHEAD * SEQ * HDIM * 2);
  _Float16* kb     = (_Float16*)carve((size_t)BATCH * NHEAD * SEQ * HDIM * 2);
  _Float16* vT     = (_Float16*)carve((size_t)BATCH * NHEAD * HDIM * SEQ * 2);
  _Float16* attn   = (_Float16*)carve((size_t)NTOK * DIMC * 2);
  float*    ybuf   = (float*)   carve((size_t)NTOK * DIMC * 4);
  _Float16* ln2b   = (_Float16*)carve((size_t)NTOK * DIMC * 2);
  _Float16* h1b    = (_Float16*)carve((size_t)NTOK * FFND * 2);

  // weight conversions
  k_f32_to_f16<<<dim3((768 * 256 + 255) / 256), 256, 0, stream>>>(
      wqkv, wqkv_h, 768 * 256);
  k_f32_to_f16<<<dim3((256 * 256 + 255) / 256), 256, 0, stream>>>(
      wproj, wprj_h, 256 * 256);
  k_f32_to_f16<<<dim3((1024 * 256 + 255) / 256), 256, 0, stream>>>(
      w1, w1_h, 1024 * 256);
  k_f32_to_f16<<<dim3((256 * 1024 + 255) / 256), 256, 0, stream>>>(
      w2, w2_h, 256 * 1024);

  k_ln1<<<dim3(NTOK / 256), 256, 0, stream>>>(x, g1, be1, xln);
  k_qkv<<<dim3(3, NTOK / 32), 256, 0, stream>>>(xln, wqkv_h, qb, kb, vT);
  k_attn<<<dim3(BATCH * NHEAD * (SEQ / 128)), 256, 0, stream>>>(qb, kb, vT,
                                                                attn);
  k_proj<<<dim3(1, NTOK / 32), 256, 0, stream>>>(attn, wprj_h, bproj, x, ybuf);
  k_ln2<<<dim3(NTOK / 8), 256, 0, stream>>>(ybuf, g2, be2, ln2b);
  k_ffn1<<<dim3(4, NTOK / 32), 256, 0, stream>>>(ln2b, w1_h, b1, h1b);
  k_ffn2<<<dim3(1, NTOK / 32), 256, 0, stream>>>(h1b, w2_h, b2, ybuf, out);
}
